// CausalSelfAttention_23768349016530
// MI455X (gfx1250) — matrix operand order repaired and checked by reading, not compiled, not run on hardware
//
#include <hip/hip_runtime.h>

#define B_ 8
#define N_ 1024
#define C_ 768
#define H_ 12
#define D_ 64
#define C3_ 2304

typedef __attribute__((ext_vector_type(16))) __bf16 v16bf;
typedef __attribute__((ext_vector_type(8))) float v8f;

struct FragBF {
  union {
    v16bf v;
    unsigned int u[8];
    unsigned short s[16];
  };
};

__device__ __forceinline__ unsigned short f2bf(float f) {
  unsigned int u = __float_as_uint(f);
  u += 0x7FFFu + ((u >> 16) & 1u);  // round-to-nearest-even
  return (unsigned short)(u >> 16);
}

__device__ __forceinline__ v8f wmma_bf16(const FragBF& a, const FragBF& b, v8f c) {
  return __builtin_amdgcn_wmma_f32_16x16x32_bf16(false, a.v, false, b.v, (short)0, c,
                                                 false, false);
}

// CDNA5 async global->LDS copy, 16 bytes per lane, tracked by ASYNCcnt.
__device__ __forceinline__ void async_b128(const unsigned short* gptr,
                                           const unsigned short* lptr) {
  unsigned lds = (unsigned)(unsigned long long)lptr;
  unsigned long long ga = (unsigned long long)gptr;
  asm volatile("global_load_async_to_lds_b128 %0, %1, off"
               :: "v"(lds), "v"(ga)
               : "memory");
}
__device__ __forceinline__ void wait_async0() {
  asm volatile("s_wait_asynccnt 0x0" ::: "memory");
}

__device__ __forceinline__ float hmax16(float x) {
  x = fmaxf(x, __shfl_xor(x, 8, 16));
  x = fmaxf(x, __shfl_xor(x, 4, 16));
  x = fmaxf(x, __shfl_xor(x, 2, 16));
  x = fmaxf(x, __shfl_xor(x, 1, 16));
  return x;
}
__device__ __forceinline__ float hsum16(float x) {
  x += __shfl_xor(x, 8, 16);
  x += __shfl_xor(x, 4, 16);
  x += __shfl_xor(x, 2, 16);
  x += __shfl_xor(x, 1, 16);
  return x;
}

// ---------------------------------------------------------------------------
// Kernel 1: fp32 -> bf16 casts. Weights stored TRANSPOSED ([NOUT][K]).
// ---------------------------------------------------------------------------
__global__ __launch_bounds__(256) void cast_kernel(
    const float* __restrict__ x, const float* __restrict__ wa,
    const float* __restrict__ wp, unsigned short* __restrict__ xb,
    unsigned short* __restrict__ wat, unsigned short* __restrict__ wpt) {
  const int tid = blockIdx.x * blockDim.x + threadIdx.x;
  const int stride = gridDim.x * blockDim.x;
  for (int i = tid; i < B_ * N_ * C_; i += stride) xb[i] = f2bf(x[i]);
  for (int i = tid; i < C3_ * C_; i += stride) {
    const int n = i / C_, kx = i - n * C_;
    wat[i] = f2bf(wa[(size_t)kx * C3_ + n]);
  }
  for (int i = tid; i < C_ * C_; i += stride) {
    const int n = i / C_, kx = i - n * C_;
    wpt[i] = f2bf(wp[(size_t)kx * C_ + n]);
  }
}

// ---------------------------------------------------------------------------
// Tiled bf16 GEMM with async-LDS, manually double-pumped double buffering.
// C[M,NOUT] = A[M,768] * Bt[NOUT,768]^T + bias
// EPI==0: scatter q/k [B,H,N,D], v transposed [B,H,D,N]. EPI==1: fp32 out.
// ---------------------------------------------------------------------------
template <int NOUT, int EPI>
__global__ __launch_bounds__(256) void gemm_bf16_kernel(
    const unsigned short* __restrict__ A, const unsigned short* __restrict__ Bt,
    const float* __restrict__ bias, float* __restrict__ outF,
    unsigned short* __restrict__ qb, unsigned short* __restrict__ kb,
    unsigned short* __restrict__ vtb) {
  constexpr int K = C_;
  constexpr int LDA_S = 40;  // 80B rows, 16B aligned
  constexpr int LDB_S = 40;
  __shared__ unsigned short As[2][128 * LDA_S];
  __shared__ unsigned short Bs[2][64 * LDB_S];

  const int tid = threadIdx.x;
  const int m0 = blockIdx.x * 128;
  const int n0 = blockIdx.y * 64;
  const int wave = tid >> 5;
  const int lane = tid & 31;
  const int wm = wave & 3;
  const int wn = wave >> 2;
  const int lane16 = lane & 15;
  const int hi = lane >> 4;
  const int khalf = hi << 3;
  const int kh16 = hi << 4;

  v8f zero = {};
  v8f acc[2][2] = {{zero, zero}, {zero, zero}};

  const int arow = tid >> 1;        // 0..127
  const int aseg = (tid & 1) << 4;  // 0/16 elements (2x b128 per thread)
  const int brow = tid >> 2;        // 0..63
  const int bseg = (tid & 3) << 3;  // 1x b128 per thread

  const unsigned short* Arow = A + (size_t)(m0 + arow) * K + aseg;
  const unsigned short* Brow = Bt + (size_t)(n0 + brow) * K + bseg;

  auto stage = [&](int kc, int nb) {
    async_b128(Arow + kc, &As[nb][arow * LDA_S + aseg]);
    async_b128(Arow + kc + 8, &As[nb][arow * LDA_S + aseg + 8]);
    async_b128(Brow + kc, &Bs[nb][brow * LDB_S + bseg]);
  };

  auto step = [&](int bufi) {
    FragBF af[2], bfm[2];
#pragma unroll
    for (int t = 0; t < 2; ++t) {
      const unsigned short* pa = &As[bufi][(wm * 32 + t * 16 + lane16) * LDA_S];
#pragma unroll
      for (int vv = 0; vv < 8; ++vv) {
        const int kk = (vv < 4 ? 2 * vv : 8 + 2 * vv) + khalf;
        af[t].u[vv] = *(const unsigned int*)(pa + kk);
      }
      const unsigned short* pb =
          &Bs[bufi][(wn * 32 + t * 16 + lane16) * LDB_S + khalf];
#pragma unroll
      for (int vv = 0; vv < 8; ++vv)
        bfm[t].u[vv] = *(const unsigned int*)(pb + (vv < 4 ? 2 * vv : 8 + 2 * vv));
    }
#pragma unroll
    for (int i = 0; i < 2; ++i)
#pragma unroll
      for (int j = 0; j < 2; ++j) acc[i][j] = wmma_bf16(af[i], bfm[j], acc[i][j]);
  };

  stage(0, 0);
#pragma unroll
  for (int kc = 0; kc < K; kc += 64) {
    wait_async0();
    __syncthreads();
    stage(kc + 32, 1);  // kc+32 <= 736 < 768 always
    step(0);
    wait_async0();
    __syncthreads();
    if (kc + 64 < K) stage(kc + 64, 0);
    step(1);
  }

#pragma unroll
  for (int i = 0; i < 2; ++i) {
#pragma unroll
    for (int j = 0; j < 2; ++j) {
#pragma unroll
      for (int r = 0; r < 8; ++r) {
        const int gm = m0 + wm * 32 + i * 16 + r + khalf;
        const int gn = n0 + wn * 32 + j * 16 + lane16;
        const float val = acc[i][j][r] + bias[gn];
        if (EPI == 1) {
          outF[(size_t)gm * NOUT + gn] = val;
        } else {
          const int bb = gm >> 10;
          const int nrow = gm & (N_ - 1);
          const int which = gn / C_;
          const int c = gn - which * C_;
          const int hh = c >> 6;
          const int dd = c & 63;
          const int bhh = bb * H_ + hh;
          if (which == 2) {
            vtb[((size_t)bhh * D_ + dd) * N_ + nrow] = f2bf(val);
          } else {
            unsigned short* dst = (which == 0) ? qb : kb;
            dst[((size_t)(bhh * N_ + nrow) << 6) + dd] = f2bf(val);
          }
        }
      }
    }
  }
}

// ---------------------------------------------------------------------------
// Kernel 3: flash-style attention. K/V tiles async-staged to LDS once per
// block (shared by all 8 waves), double-pumped double buffering.
// ---------------------------------------------------------------------------
__global__ __launch_bounds__(256) void attn_kernel(
    const unsigned short* __restrict__ q, const unsigned short* __restrict__ k,
    const unsigned short* __restrict__ vt, const int* __restrict__ pad,
    unsigned short* __restrict__ o) {
  constexpr int LDK_S = 72;  // [key][d] rows: 144B, 16B aligned
  constexpr int LDV_S = 40;  // [d][key] rows: 80B, 16B aligned
  __shared__ unsigned short Ks[2][32 * LDK_S];
  __shared__ unsigned short Vs[2][64 * LDV_S];
  __shared__ unsigned short Pl[8][16][36];

  const int blk = blockIdx.x;
  const int bh = blk >> 3;
  const int chunk = blk & 7;
  const int bb = bh / H_;
  const int hh = bh - bb * H_;
  const int tid = threadIdx.x;
  const int wave = tid >> 5;
  const int lane = tid & 31;
  const int lane16 = lane & 15;
  const int hi = lane >> 4;
  const int khalf = hi << 3;
  const int kh16 = hi << 4;
  const int q0 = chunk * 128 + wave * 16;

  const unsigned short* Q = q + (size_t)bh * N_ * D_;
  const unsigned short* Kp = k + (size_t)bh * N_ * D_;
  const unsigned short* Vt = vt + (size_t)bh * D_ * N_;
  unsigned short(*Pw)[36] = Pl[wave];

  const int krow = tid >> 3, kseg = (tid & 7) << 3;  // K tile 32x64
  const int vrow = tid >> 2, vseg = (tid & 3) << 3;  // V tile 64x32 (transposed)
  const unsigned short* Kg = Kp + (size_t)krow * D_ + kseg;
  const unsigned short* Vg = Vt + (size_t)vrow * N_ + vseg;

  FragBF qf[2];
#pragma unroll
  for (int c = 0; c < 2; ++c) {
    const unsigned short* qp = Q + (q0 + lane16) * D_ + c * 32;
#pragma unroll
    for (int vv = 0; vv < 8; ++vv) {
      const int kk = (vv < 4 ? 2 * vv : 8 + 2 * vv) + khalf;
      qf[c].u[vv] = *(const unsigned int*)(qp + kk);
    }
  }

  float rowbias[8];
#pragma unroll
  for (int r = 0; r < 8; ++r)
    rowbias[r] = pad[bb * N_ + q0 + r + khalf] ? 0.0f : -1e30f;

  float m8[8], l8[8], a8[8];
  v8f zero = {};
  v8f oacc[4] = {zero, zero, zero, zero};
#pragma unroll
  for (int r = 0; r < 8; ++r) {
    m8[r] = -3.0e38f;
    l8[r] = 0.0f;
  }

  auto stageKV = [&](int kt, int nb) {
    async_b128(Kg + (size_t)kt * D_, &Ks[nb][krow * LDK_S + kseg]);
    async_b128(Vg + kt, &Vs[nb][vrow * LDV_S + vseg]);
  };

  const float scale = 0.125f;  // 1/sqrt(64)

  auto astep = [&](const unsigned short* Kb, const unsigned short* Vb) {
    // --- preload all four K fragments, then issue the 4 S-WMMAs ---
    FragBF kf00, kf01, kf10, kf11;  // [key-subtile][d-chunk]
    const unsigned short* kb0 = Kb + lane16 * LDK_S + khalf;
    const unsigned short* kb1 = Kb + (16 + lane16) * LDK_S + khalf;
#pragma unroll
    for (int vv = 0; vv < 8; ++vv) {
      const int ko = vv < 4 ? 2 * vv : 8 + 2 * vv;
      kf00.u[vv] = *(const unsigned int*)(kb0 + ko);
      kf01.u[vv] = *(const unsigned int*)(kb0 + 32 + ko);
      kf10.u[vv] = *(const unsigned int*)(kb1 + ko);
      kf11.u[vv] = *(const unsigned int*)(kb1 + 32 + ko);
    }
    v8f s0 = zero, s1 = zero;
    s0 = wmma_bf16(qf[0], kf00, s0);
    s1 = wmma_bf16(qf[0], kf10, s1);
    s0 = wmma_bf16(qf[1], kf01, s0);
    s1 = wmma_bf16(qf[1], kf11, s1);

    // --- preload V fragments early (independent of softmax/P staging) ---
    FragBF vf[4];
#pragma unroll
    for (int dt = 0; dt < 4; ++dt) {
      const unsigned short* pv = Vb + (dt * 16 + lane16) * LDV_S + khalf;
#pragma unroll
      for (int vv = 0; vv < 8; ++vv)
        vf[dt].u[vv] = *(const unsigned int*)(pv + (vv < 4 ? 2 * vv : 8 + 2 * vv));
    }

    // --- online softmax (row M = r + 8*hi, one 16-lane half per row) ---
#pragma unroll
    for (int r = 0; r < 8; ++r) {
      const float x0 = s0[r] * scale + rowbias[r];
      const float x1 = s1[r] * scale + rowbias[r];
      const float mx = hmax16(fmaxf(x0, x1));
      const float mnew = fmaxf(m8[r], mx);
      const float p0 = __expf(x0 - mnew);
      const float p1 = __expf(x1 - mnew);
      const float rs = hsum16(p0 + p1);
      const float alpha = __expf(m8[r] - mnew);
      l8[r] = l8[r] * alpha + rs;
      m8[r] = mnew;
      a8[r] = alpha;
      Pw[r + khalf][lane16] = f2bf(p0);
      Pw[r + khalf][16 + lane16] = f2bf(p1);
    }
#pragma unroll
    for (int dt = 0; dt < 4; ++dt)
#pragma unroll
      for (int r = 0; r < 8; ++r) oacc[dt][r] *= a8[r];

    // --- P as A-fragment from per-wave LDS, then 4 O-WMMAs back-to-back ---
    FragBF pf;
    const unsigned short* pp = &Pw[lane16][0];
#pragma unroll
    for (int vv = 0; vv < 8; ++vv) {
      const int kk = (vv < 4 ? 2 * vv : 8 + 2 * vv) + khalf;
      pf.u[vv] = *(const unsigned int*)(pp + kk);
    }
#pragma unroll
    for (int dt = 0; dt < 4; ++dt) oacc[dt] = wmma_bf16(pf, vf[dt], oacc[dt]);
  };

  stageKV(0, 0);
#pragma unroll 2
  for (int kt = 0; kt < N_; kt += 64) {
    wait_async0();
    __syncthreads();
    stageKV(kt + 32, 1);  // kt+32 <= 992 < 1024 always
    astep(&Ks[0][0], &Vs[0][0]);
    wait_async0();
    __syncthreads();
    if (kt + 64 < N_) stageKV(kt + 64, 0);
    astep(&Ks[1][0], &Vs[1][0]);
  }

#pragma unroll
  for (int dt = 0; dt < 4; ++dt) {
#pragma unroll
    for (int r = 0; r < 8; ++r) {
      const float val = oacc[dt][r] / l8[r];
      const int qrow = q0 + r + khalf;
      const int col = hh * D_ + dt * 16 + lane16;
      o[(size_t)(bb * N_ + qrow) * C_ + col] = f2bf(val);
    }
  }
}

// ---------------------------------------------------------------------------
// Launch
// ---------------------------------------------------------------------------
extern "C" void kernel_launch(void* const* d_in, const int* in_sizes, int n_in,
                              void* d_out, int out_size, void* d_ws,
                              size_t ws_size, hipStream_t stream) {
  const float* x = (const float*)d_in[0];
  const int* pad = (const int*)d_in[1];
  const float* w_attn = (const float*)d_in[2];
  const float* b_attn = (const float*)d_in[3];
  const float* w_proj = (const float*)d_in[4];
  const float* b_proj = (const float*)d_in[5];
  float* out = (float*)d_out;

  char* ws = (char*)d_ws;
  constexpr size_t SZ_XB = (size_t)B_ * N_ * C_ * 2;
  constexpr size_t SZ_WA = (size_t)C_ * C3_ * 2;
  constexpr size_t SZ_WP = (size_t)C_ * C_ * 2;
  constexpr size_t SZ_HD = (size_t)B_ * H_ * N_ * D_ * 2;

  unsigned short* xb = (unsigned short*)(ws);
  unsigned short* wat = (unsigned short*)(ws + SZ_XB);
  unsigned short* wpt = (unsigned short*)(ws + SZ_XB + SZ_WA);
  unsigned short* qb = (unsigned short*)(ws + SZ_XB + SZ_WA + SZ_WP);
  unsigned short* kb = (unsigned short*)(ws + SZ_XB + SZ_WA + SZ_WP + SZ_HD);
  unsigned short* vtb = (unsigned short*)(ws + SZ_XB + SZ_WA + SZ_WP + 2 * SZ_HD);
  unsigned short* ab = (unsigned short*)(ws + SZ_XB + SZ_WA + SZ_WP + 3 * SZ_HD);

  cast_kernel<<<8192, 256, 0, stream>>>(x, w_attn, w_proj, xb, wat, wpt);

  gemm_bf16_kernel<C3_, 0><<<dim3(64, 36), 256, 0, stream>>>(
      xb, wat, b_attn, nullptr, qb, kb, vtb);

  attn_kernel<<<768, 256, 0, stream>>>(qb, kb, vtb, pad, ab);

  gemm_bf16_kernel<C_, 1><<<dim3(64, 12), 256, 0, stream>>>(
      ab, wpt, b_proj, out, nullptr, nullptr, nullptr);
}
